// RoPEAttention_9818295238902
// MI455X (gfx1250) — matrix operand order repaired and checked by reading, not compiled, not run on hardware
//
#include <hip/hip_runtime.h>
#include <hip/hip_bf16.h>

typedef __attribute__((ext_vector_type(16))) _Float16 v16h;
typedef __attribute__((ext_vector_type(8)))  float    v8f;

#define B_SZ   2
#define S_LEN  2048
#define DM     1024
#define NH     16
#define HD     64
#define EPS    1e-6f

static __device__ __forceinline__ v8f wmma_f16(v16h a, v16h b, v8f c) {
    return __builtin_amdgcn_wmma_f32_16x16x32_f16(false, a, false, b, (short)0, c, false, false);
}

// ---------------------------------------------------------------------------
// Stage 0: one-shot fp32 -> f16 materialization (removes all cvt from hot loops)
// ---------------------------------------------------------------------------
__global__ __launch_bounds__(256) void cvt_f16_kernel(
    const float* __restrict__ src, _Float16* __restrict__ dst, int n)
{
    const int i = (blockIdx.x * 256 + threadIdx.x) * 8;
    if (i < n) {
        #pragma unroll
        for (int j = 0; j < 8; ++j) dst[i + j] = (_Float16)src[i + j];
    }
}

// ---------------------------------------------------------------------------
// Stage 1: q/k/v = x @ W^T  (WMMA f16, fp32 accum), then RMSNorm + RoPE (fp32)
// blockIdx: x = row-tile group (8 waves x 16 rows), y = head, z = which (0=q,1=k,2=v)
// Each wave computes a 16x64 tile (one full head) so norm is intra-wave.
// ---------------------------------------------------------------------------
__global__ __launch_bounds__(256) void qkv_rope_kernel(
    const _Float16* __restrict__ xh,
    const _Float16* __restrict__ wqh, const _Float16* __restrict__ wkh,
    const _Float16* __restrict__ wvh,
    const float* __restrict__ q_scale, const float* __restrict__ k_scale,
    const float* __restrict__ rope_cos, const float* __restrict__ rope_sin,
    _Float16* __restrict__ q_out, _Float16* __restrict__ k_out, _Float16* __restrict__ v_out)
{
    const int which = blockIdx.z;
    const _Float16* __restrict__ w = (which == 0) ? wqh : (which == 1) ? wkh : wvh;
    const int h    = blockIdx.y;
    const int wave = threadIdx.x >> 5;
    const int lane = threadIdx.x & 31;
    const int m    = lane & 15;
    const int hi   = lane >> 4;
    const int row0 = (blockIdx.x * 8 + wave) * 16;

    const v8f vzero = {0.f,0.f,0.f,0.f,0.f,0.f,0.f,0.f};
    v8f acc[4];
    #pragma unroll
    for (int t = 0; t < 4; ++t) acc[t] = vzero;

    const _Float16* __restrict__ xrow = xh + (size_t)(row0 + m) * DM;

    for (int ks = 0; ks < DM / 32; ++ks) {
        const int kbase = ks * 32;
        // A fragment: 16x32 f16, lane m = row; k = 16*hi + {0..15} (contig)
        v16h a;
        const _Float16* ap = xrow + kbase + 16 * hi;
        #pragma unroll
        for (int j = 0; j < 8; ++j) a[j]     = ap[j];
        #pragma unroll
        for (int j = 0; j < 8; ++j) a[8 + j] = ap[8 + j];
        #pragma unroll
        for (int t = 0; t < 4; ++t) {
            // B fragment: lane n = output col = h*64 + 16t + m; k = 16*hi + j (contig)
            const _Float16* bp = w + (size_t)(h * HD + t * 16 + m) * DM + kbase + 16 * hi;
            v16h bf;
            #pragma unroll
            for (int j = 0; j < 16; ++j) bf[j] = bp[j];
            acc[t] = wmma_f16(a, bf, acc[t]);
        }
    }

    if (which < 2) {
        const float* __restrict__ sc = (which == 0) ? q_scale : k_scale;
        float scl[4];
        #pragma unroll
        for (int t = 0; t < 4; ++t) scl[t] = sc[t * 16 + m];

        // RMSNorm per row (row = g + 8*hi lives across lanes 0-15 / 16-31)
        float rstd[8];
        #pragma unroll
        for (int g = 0; g < 8; ++g) {
            float ss = 0.f;
            #pragma unroll
            for (int t = 0; t < 4; ++t) { float v = acc[t][g]; ss += v * v; }
            #pragma unroll
            for (int off = 1; off < 16; off <<= 1) ss += __shfl_xor(ss, off, 32);
            rstd[g] = rsqrtf(ss * (1.0f / (float)HD) + EPS);
        }
        #pragma unroll
        for (int t = 0; t < 4; ++t)
            #pragma unroll
            for (int g = 0; g < 8; ++g) acc[t][g] *= rstd[g] * scl[t];

        const float qmul = (which == 0) ? 0.125f : 1.0f;  // fold 1/sqrt(HD) into Q
        _Float16* __restrict__ outp = (which == 0) ? q_out : k_out;
        #pragma unroll
        for (int g = 0; g < 8; ++g) {
            const int r  = row0 + g + 8 * hi;
            const int b  = r >> 11;          // S = 2048
            const int sp = r & (S_LEN - 1);
            const float* cr = rope_cos + (size_t)sp * (HD / 2);
            const float* sr = rope_sin + (size_t)sp * (HD / 2);
            #pragma unroll
            for (int t = 0; t < 4; ++t) {
                const float c  = cr[t * 8 + (m >> 1)];
                const float sn = sr[t * 8 + (m >> 1)];
                const float v  = acc[t][g];
                const float pv = __shfl_xor(v, 1, 32);  // partner of the rope pair
                float o = (m & 1) ? (pv * sn + v * c) : (v * c - pv * sn);
                o *= qmul;
                const int hd = t * 16 + m;
                outp[((size_t)(b * NH + h) * S_LEN + sp) * HD + hd] = (_Float16)o;
            }
        }
    } else {
        // V: no norm/rope; store TRANSPOSED [bh][hd][s] for stage-2 B fragments
        #pragma unroll
        for (int g = 0; g < 8; ++g) {
            const int r  = row0 + g + 8 * hi;
            const int b  = r >> 11;
            const int sp = r & (S_LEN - 1);
            #pragma unroll
            for (int t = 0; t < 4; ++t) {
                const int hd = t * 16 + m;
                v_out[((size_t)(b * NH + h) * HD + hd) * S_LEN + sp] = (_Float16)acc[t][g];
            }
        }
    }
}

// ---------------------------------------------------------------------------
// Stage 2: flash attention. blockIdx: x = q-row tile group, y = b*H + h.
// Each wave owns 16 q rows; online softmax fully intra-wave (16-lane shfl).
// ---------------------------------------------------------------------------
__global__ __launch_bounds__(256) void attn_kernel(
    const _Float16* __restrict__ q_ws,   // [bh][s][hd], pre-scaled by 0.125
    const _Float16* __restrict__ k_ws,   // [bh][s][hd]
    const _Float16* __restrict__ v_ws,   // [bh][hd][s]  (transposed)
    _Float16* __restrict__ a_ws)         // [b][s][h][hd]
{
    __shared__ __align__(16) _Float16 plds[8][16][40];   // per-wave P bounce (C->A relayout)

    const int bh   = blockIdx.y;
    const int wave = threadIdx.x >> 5;
    const int lane = threadIdx.x & 31;
    const int m    = lane & 15;
    const int hi   = lane >> 4;
    const int q0   = (blockIdx.x * 8 + wave) * 16;

    const _Float16* __restrict__ qb = q_ws + (size_t)bh * S_LEN * HD;
    const _Float16* __restrict__ kb = k_ws + (size_t)bh * S_LEN * HD;
    const _Float16* __restrict__ vb = v_ws + (size_t)bh * HD * S_LEN;

    // Q fragments (A-layout), one per 32-wide k-step over hd
    v16h qf[2];
    #pragma unroll
    for (int c = 0; c < 2; ++c) {
        const _Float16* qp = qb + (size_t)(q0 + m) * HD + c * 32 + 16 * hi;
        #pragma unroll
        for (int j = 0; j < 8; ++j) qf[c][j]     = qp[j];
        #pragma unroll
        for (int j = 0; j < 8; ++j) qf[c][8 + j] = qp[8 + j];
    }

    const v8f vzero = {0.f,0.f,0.f,0.f,0.f,0.f,0.f,0.f};
    v8f o[4];
    #pragma unroll
    for (int t = 0; t < 4; ++t) o[t] = vzero;
    float mrun[8], lrun[8];
    #pragma unroll
    for (int g = 0; g < 8; ++g) { mrun[g] = -3.0e38f; lrun[g] = 0.f; }

    for (int kb0 = 0; kb0 < S_LEN; kb0 += 32) {
        // scores: 16 q-rows x 32 keys (two 16x16 C tiles), K-dim = hd = 64
        v8f s0 = vzero, s1 = vzero;
        #pragma unroll
        for (int c = 0; c < 2; ++c) {
            v16h kf0, kf1;
            const _Float16* kp0 = kb + (size_t)(kb0 + m) * HD + c * 32 + 16 * hi;
            const _Float16* kp1 = kb + (size_t)(kb0 + 16 + m) * HD + c * 32 + 16 * hi;
            #pragma unroll
            for (int j = 0; j < 16; ++j) kf0[j] = kp0[j];
            #pragma unroll
            for (int j = 0; j < 16; ++j) kf1[j] = kp1[j];
            s0 = wmma_f16(qf[c], kf0, s0);
            s1 = wmma_f16(qf[c], kf1, s1);
        }

        // online softmax (fp32)
        float p0[8], p1[8];
        #pragma unroll
        for (int g = 0; g < 8; ++g) {
            float rm = fmaxf(s0[g], s1[g]);
            #pragma unroll
            for (int off = 1; off < 16; off <<= 1) rm = fmaxf(rm, __shfl_xor(rm, off, 32));
            const float mn = fmaxf(mrun[g], rm);
            const float al = __expf(mrun[g] - mn);
            const float e0 = __expf(s0[g] - mn);
            const float e1 = __expf(s1[g] - mn);
            float rs = e0 + e1;
            #pragma unroll
            for (int off = 1; off < 16; off <<= 1) rs += __shfl_xor(rs, off, 32);
            lrun[g] = lrun[g] * al + rs;
            mrun[g] = mn;
            o[0][g] *= al; o[1][g] *= al; o[2][g] *= al; o[3][g] *= al;
            p0[g] = e0; p1[g] = e1;
        }

        // stage P (C layout) into LDS, read back in A layout
        #pragma unroll
        for (int g = 0; g < 8; ++g) {
            const int r = g + 8 * hi;
            plds[wave][r][m]      = (_Float16)p0[g];
            plds[wave][r][16 + m] = (_Float16)p1[g];
        }
        __syncthreads();
        v16h pf;
        {
            const _Float16* pp = &plds[wave][m][16 * hi];
            #pragma unroll
            for (int j = 0; j < 8; ++j) pf[j]     = pp[j];
            #pragma unroll
            for (int j = 0; j < 8; ++j) pf[8 + j] = pp[8 + j];
        }
        __syncthreads();

        // O += P @ V   (K-dim = 32 keys; B fragment contiguous thanks to V^T)
        #pragma unroll
        for (int t = 0; t < 4; ++t) {
            const _Float16* vp = vb + (size_t)(t * 16 + m) * S_LEN + kb0 + 16 * hi;
            v16h vf;
            #pragma unroll
            for (int j = 0; j < 16; ++j) vf[j] = vp[j];
            o[t] = wmma_f16(pf, vf, o[t]);
        }
    }

    // normalize and store attn as [b][s][h][hd] (k-major for the final GEMM)
    const int b = bh >> 4, h = bh & 15;
    #pragma unroll
    for (int g = 0; g < 8; ++g) {
        const float inv = 1.0f / lrun[g];
        const int sp = q0 + g + 8 * hi;
        #pragma unroll
        for (int t = 0; t < 4; ++t) {
            const int hd = t * 16 + m;
            a_ws[((size_t)(b * S_LEN + sp) * NH + h) * HD + hd] = (_Float16)(o[t][g] * inv);
        }
    }
}

// ---------------------------------------------------------------------------
// Stage 3: out = attn @ wo^T (fp32 out). blockIdx: x = row tiles, y = col tiles.
// ---------------------------------------------------------------------------
__global__ __launch_bounds__(256) void out_proj_kernel(
    const _Float16* __restrict__ a_ws, const _Float16* __restrict__ woh,
    float* __restrict__ out)
{
    const int wave = threadIdx.x >> 5;
    const int lane = threadIdx.x & 31;
    const int m    = lane & 15;
    const int hi   = lane >> 4;
    const int row0 = (blockIdx.x * 8 + wave) * 16;
    const int n0   = blockIdx.y * 64;

    const v8f vzero = {0.f,0.f,0.f,0.f,0.f,0.f,0.f,0.f};
    v8f acc[4];
    #pragma unroll
    for (int t = 0; t < 4; ++t) acc[t] = vzero;

    const _Float16* __restrict__ arow = a_ws + (size_t)(row0 + m) * DM;
    for (int ks = 0; ks < DM / 32; ++ks) {
        const int kbase = ks * 32;
        v16h a;
        const _Float16* ap = arow + kbase + 16 * hi;
        #pragma unroll
        for (int j = 0; j < 8; ++j) a[j]     = ap[j];
        #pragma unroll
        for (int j = 0; j < 8; ++j) a[8 + j] = ap[8 + j];
        #pragma unroll
        for (int t = 0; t < 4; ++t) {
            const _Float16* bp = woh + (size_t)(n0 + t * 16 + m) * DM + kbase + 16 * hi;
            v16h bf;
            #pragma unroll
            for (int j = 0; j < 16; ++j) bf[j] = bp[j];
            acc[t] = wmma_f16(a, bf, acc[t]);
        }
    }
    #pragma unroll
    for (int g = 0; g < 8; ++g) {
        const int r = row0 + g + 8 * hi;
        #pragma unroll
        for (int t = 0; t < 4; ++t)
            out[(size_t)r * DM + n0 + t * 16 + m] = acc[t][g];
    }
}

extern "C" void kernel_launch(void* const* d_in, const int* in_sizes, int n_in,
                              void* d_out, int out_size, void* d_ws, size_t ws_size,
                              hipStream_t stream) {
    const float* x        = (const float*)d_in[0];
    const float* wq       = (const float*)d_in[1];
    const float* wk       = (const float*)d_in[2];
    const float* wv       = (const float*)d_in[3];
    const float* wo       = (const float*)d_in[4];
    const float* q_scale  = (const float*)d_in[5];
    const float* k_scale  = (const float*)d_in[6];
    const float* rope_cos = (const float*)d_in[7];
    const float* rope_sin = (const float*)d_in[8];
    float* out = (float*)d_out;

    const size_t qkv_elems = (size_t)B_SZ * NH * S_LEN * HD;  // 4M f16 elems
    const size_t x_elems   = (size_t)B_SZ * S_LEN * DM;       // 4M
    const size_t w_elems   = (size_t)DM * DM;                 // 1M

    _Float16* q_ws = (_Float16*)d_ws;
    _Float16* k_ws = q_ws + qkv_elems;
    _Float16* v_ws = k_ws + qkv_elems;
    _Float16* a_ws = v_ws + qkv_elems;
    _Float16* xh   = a_ws + qkv_elems;
    _Float16* wqh  = xh  + x_elems;
    _Float16* wkh  = wqh + w_elems;
    _Float16* wvh  = wkh + w_elems;
    _Float16* woh  = wvh + w_elems;

    // Stage 0: materialize f16 copies of x and the four weight matrices
    cvt_f16_kernel<<<dim3((int)(x_elems / (8 * 256))), 256, 0, stream>>>(x,  xh,  (int)x_elems);
    cvt_f16_kernel<<<dim3((int)(w_elems / (8 * 256))), 256, 0, stream>>>(wq, wqh, (int)w_elems);
    cvt_f16_kernel<<<dim3((int)(w_elems / (8 * 256))), 256, 0, stream>>>(wk, wkh, (int)w_elems);
    cvt_f16_kernel<<<dim3((int)(w_elems / (8 * 256))), 256, 0, stream>>>(wv, wvh, (int)w_elems);
    cvt_f16_kernel<<<dim3((int)(w_elems / (8 * 256))), 256, 0, stream>>>(wo, woh, (int)w_elems);

    qkv_rope_kernel<<<dim3(32, NH, 3), 256, 0, stream>>>(
        xh, wqh, wkh, wvh, q_scale, k_scale, rope_cos, rope_sin, q_ws, k_ws, v_ws);
    attn_kernel<<<dim3(16, B_SZ * NH), 256, 0, stream>>>(q_ws, k_ws, v_ws, a_ws);
    out_proj_kernel<<<dim3(32, DM / 64), 256, 0, stream>>>(a_ws, woh, out);
}
